// D2Layer_32246614458526
// MI455X (gfx1250) — matrix operand order repaired and checked by reading, not compiled, not run on hardware
//
#include <hip/hip_runtime.h>

// ---------------------------------------------------------------------------
// Fused discretize + 7-layer MLP for MI455X (gfx1250), wave32 + WMMA f16,
// with Tensor-Data-Mover (TDM) double-buffered weight staging.
//   x:[131072,512] f32 -> per-column 25-level discretize -> f16
//   h = relu(d @ W_in + b_in); 5x h = relu(h @ W_h + b_h); y = relu(h @ W_out + b_out)
// Activations stay in LDS (f16, double-buffered) across all layers.
// LDS layout puts the weight panels at offset 0 so B-fragment ds_loads can
// fold their addresses into the 16-bit DS immediate-offset field.
// ---------------------------------------------------------------------------

typedef _Float16 h4_t  __attribute__((ext_vector_type(4)));
typedef _Float16 h8_t  __attribute__((ext_vector_type(8)));
typedef _Float16 v16h  __attribute__((ext_vector_type(16)));
typedef float    v8f   __attribute__((ext_vector_type(8)));
typedef unsigned int u32x4 __attribute__((ext_vector_type(4)));
typedef int      i32x4 __attribute__((ext_vector_type(4)));
typedef int      i32x8 __attribute__((ext_vector_type(8)));

#define HID    512          // hidden width (= D_in)
#define DOUT   128          // output width
#define BTOT   131072       // batch
#define MB     128          // rows per block
#define PADH   8            // f16 padding per LDS row
#define LDH    (HID + PADH) // 520 halves = 1040 B row stride (16B aligned, 4-bank shift)
#define NBLK   (BTOT / MB)  // 1024 blocks
#define LEVELS 25
#define PANELH (16 * LDH)   // one weight panel, in halves

// ---- order-preserving float<->uint map for atomic min/max --------------------
__device__ __forceinline__ unsigned fenc(float f) {
  unsigned u = __float_as_uint(f);
  return (u & 0x80000000u) ? ~u : (u | 0x80000000u);
}
__device__ __forceinline__ float fdec(unsigned u) {
  unsigned v = (u & 0x80000000u) ? (u & 0x7fffffffu) : ~u;
  return __uint_as_float(v);
}

// ---- K0: init min/max accumulators + d2_loss scalar --------------------------
__global__ __launch_bounds__(512) void init_minmax(unsigned* __restrict__ umin,
                                                   unsigned* __restrict__ umax,
                                                   float* __restrict__ d2loss) {
  int i = threadIdx.x;
  umin[i] = 0xFFFFFFFFu;
  umax[i] = 0u;
  if (i == 0) *d2loss = 0.0f;
}

// ---- K1: per-column min/max over all rows (coalesced, 1 thread = 1 column) ---
__global__ __launch_bounds__(512) void colminmax(const float* __restrict__ x,
                                                 unsigned* __restrict__ umin,
                                                 unsigned* __restrict__ umax) {
  const int c = threadIdx.x;
  const float* xp = x + (size_t)blockIdx.x * 256 * HID + c;
  unsigned lmin = 0xFFFFFFFFu, lmax = 0u;
#pragma unroll 4
  for (int r = 0; r < 256; ++r) {
    unsigned e = fenc(xp[(size_t)r * HID]);
    lmin = e < lmin ? e : lmin;
    lmax = e > lmax ? e : lmax;
  }
  atomicMin(&umin[c], lmin);
  atomicMax(&umax[c], lmax);
}

// ---- K2: W[k][n] f32 -> Wt[n][k] f16 (transposed for contiguous-K B frags) ---
__global__ __launch_bounds__(256) void convert_weights(const float* __restrict__ W,
                                                       _Float16* __restrict__ Wt,
                                                       int K, int N) {
  int total = K * N;
  for (int i = blockIdx.x * blockDim.x + threadIdx.x; i < total;
       i += gridDim.x * blockDim.x) {
    int n = i / K;
    int k = i - n * K;
    Wt[i] = (_Float16)W[(size_t)k * N + n];
  }
}

// ---- TDM: DMA one transposed-weight panel (16 rows x 512 f16) into LDS -------
// Pads each 1024 B row with 16 B so the LDS image has stride LDH (=1040 B):
//   pad_interval = 7 -> pad every 256 DWORDs (1024 B)
//   pad_amount   = 3 -> insert 4 DWORDs (16 B)
__device__ __forceinline__ unsigned lds_off(const void* p) {
  return (unsigned)reinterpret_cast<unsigned long long>(p);  // LDS addr = low 32 bits
}

__device__ __forceinline__ void tdm_load_panel(const _Float16* gsrc, unsigned ldsAddr) {
  unsigned long long ga = reinterpret_cast<unsigned long long>(gsrc);
  u32x4 g0;
  g0[0] = 1u;                                     // count=1, user descriptor
  g0[1] = ldsAddr;                                // lds_addr (bytes)
  g0[2] = (unsigned)(ga & 0xffffffffu);           // global_addr[31:0]
  g0[3] = (unsigned)((ga >> 32) & 0x01ffffffu)    // global_addr[56:32]
          | (2u << 30);                           // type=2 ("image")
  i32x8 g1;
  g1[0] = (int)((1u << 16)                        // data_size: 2 bytes
                | (1u << 20)                      // pad_enable
                | (7u << 22)                      // pad_interval: 256 DWORDs
                | (3u << 25));                    // pad_amount: 4 DWORDs
  g1[1] = (int)((unsigned)HID << 16);             // ab_addr=0 | tensor_dim0[15:0]=512
  g1[2] = (int)((unsigned)16 << 16);              // tensor_dim0[31:16]=0 | tensor_dim1=16
  g1[3] = (int)((unsigned)HID << 16);             // tensor_dim1[31:16]=0 | tile_dim0=512
  g1[4] = 16;                                     // tile_dim1=16 | tile_dim2=0
  g1[5] = HID;                                    // tensor_dim0_stride[31:0]=512
  g1[6] = 0;                                      // stride0 hi | tensor_dim1_stride lo
  g1[7] = 0;                                      // tensor_dim1_stride hi
  i32x4 z4 = {0, 0, 0, 0};
#if __clang_major__ >= 23
  i32x8 z8 = {0, 0, 0, 0, 0, 0, 0, 0};
  __builtin_amdgcn_tensor_load_to_lds(g0, g1, z4, z4, z8, 0);
#else
  __builtin_amdgcn_tensor_load_to_lds(g0, g1, z4, z4, 0);
#endif
}

// ---- one dense layer: hD = relu(hS @ W + b), via V_WMMA_F32_16X16X32_F16 -----
// hS/hD: [MB][LDH] f16 in LDS. Wave w owns rows w*16..w*16+15 (A frags are
// wave-private). Weight panels (16 cols x 512 K) are TDM-DMA'd by wave 0 into
// the double-buffered staging area at LDS offset 0, one panel ahead of
// compute. Panel select is an integer offset from one LDS base so the backend
// keeps DS addressing and can fold offsets into ds_load immediates.
__device__ __forceinline__ void mlp_layer(const _Float16* __restrict__ hS,
                                          _Float16* __restrict__ hD,
                                          _Float16* __restrict__ sB0,
                                          const _Float16* __restrict__ Wt,
                                          const float* __restrict__ bias,
                                          int ncols,
                                          float* __restrict__ gout,
                                          int rowBase) {
  const int tid    = threadIdx.x;
  const int lane   = tid & 31;
  const int wave   = tid >> 5;
  const int laneN  = lane & 15;   // N (and A-row select)
  const int laneHi = lane >> 4;   // K-half select / C-row offset
  const int m0     = wave * 16;

  const int ntiles = ncols >> 4;
  const unsigned sbBase = lds_off(sB0);

  if (wave == 0)  // prologue: DMA panel 0
    tdm_load_panel(Wt, sbBase);

  for (int nt = 0; nt < ntiles; ++nt) {
    if (wave == 0) {
      if (nt + 1 < ntiles) {
        // prefetch next panel into the other half (freed by the barrier at
        // the end of iteration nt-1), then wait for panel nt (in-order TDM).
        tdm_load_panel(Wt + (size_t)(nt + 1) * 16 * HID,
                       sbBase + (unsigned)(((nt + 1) & 1) * PANELH * 2));
        __builtin_amdgcn_s_wait_tensorcnt(1);
      } else {
        __builtin_amdgcn_s_wait_tensorcnt(0);
      }
    }
    __syncthreads();  // panel nt visible to all waves
    const _Float16* sB = sB0 + (nt & 1) * PANELH;  // offset select keeps DS path

    // accumulator preloaded with bias: acc[r] = C[M = r + 8*laneHi][N = laneN]
    float bv = bias[nt * 16 + laneN];
    v8f acc = {bv, bv, bv, bv, bv, bv, bv, bv};

    // A frag (16x32 f16): lane<16 -> [K0..7,K16..23], lane>=16 -> [K8..15,K24..31]
    const _Float16* aP = hS + (m0 + laneN) * LDH + laneHi * 8;
    // B frag (32x16 f16): lane holds column laneN, lane<16 -> [K0..7,K16..23], lane>=16 -> [K8..15,K24..31]
    const _Float16* bP = sB + laneN * LDH + laneHi * 8;

#pragma unroll 4
    for (int kk = 0; kk < HID; kk += 32) {
      h8_t alo = *(const h8_t*)(aP + kk);
      h8_t ahi = *(const h8_t*)(aP + kk + 16);
      h8_t blo = *(const h8_t*)(bP + kk);
      h8_t bhi = *(const h8_t*)(bP + kk + 16);
      v16h Am = __builtin_shufflevector(alo, ahi, 0, 1, 2, 3, 4, 5, 6, 7,
                                        8, 9, 10, 11, 12, 13, 14, 15);
      v16h Bm = __builtin_shufflevector(blo, bhi, 0, 1, 2, 3, 4, 5, 6, 7,
                                        8, 9, 10, 11, 12, 13, 14, 15);
      acc = __builtin_amdgcn_wmma_f32_16x16x32_f16(
          /*neg_a=*/false, Am, /*neg_b=*/false, Bm,
          /*c_mod=*/(short)0, acc, /*reuse_a=*/false, /*reuse_b=*/false);
    }

    if (gout) {  // final layer: relu -> f32 global store
#pragma unroll
      for (int r = 0; r < 8; ++r) {
        float v = acc[r] > 0.0f ? acc[r] : 0.0f;
        gout[(size_t)(rowBase + m0 + laneHi * 8 + r) * DOUT + nt * 16 + laneN] = v;
      }
    } else {  // hidden layer: relu -> f16 LDS store
#pragma unroll
      for (int r = 0; r < 8; ++r) {
        float v = acc[r] > 0.0f ? acc[r] : 0.0f;
        hD[(m0 + laneHi * 8 + r) * LDH + nt * 16 + laneN] = (_Float16)v;
      }
    }
    __syncthreads();  // all waves done reading panel nt; buffer reusable
  }
}

// ---- K3: fused discretize + 7 layers, activations resident in LDS ------------
__global__ __launch_bounds__(256) void fused_mlp(
    const float* __restrict__ x,
    const unsigned* __restrict__ umin, const unsigned* __restrict__ umax,
    const _Float16* __restrict__ WtIn, const _Float16* __restrict__ WtH,
    const _Float16* __restrict__ WtOut,
    const float* __restrict__ bIn, const float* __restrict__ bH,
    const float* __restrict__ bOut,
    float* __restrict__ y) {
  extern __shared__ _Float16 smem[];
  // Weight panels FIRST (LDS offset 0) so B-frag ds_loads use small immediates.
  _Float16* sB0 = smem;                          // 2x [16][LDH] weight panels
  float*    cmn = (float*)(sB0 + 2 * PANELH);    // [HID] column min
  float*    csc = cmn + HID;                     // [HID] column scale
  _Float16* h0  = (_Float16*)(csc + HID);        // [MB][LDH]
  _Float16* h1  = h0 + MB * LDH;                 // [MB][LDH]

  const int tid     = threadIdx.x;
  const int rowBase = blockIdx.x * MB;

  for (int c = tid; c < HID; c += 256) {
    float mn  = fdec(umin[c]);
    float mx  = fdec(umax[c]);
    float rng = mx - mn;
    cmn[c] = mn;
    csc[c] = rng > 0.0f ? (float)LEVELS / rng : 0.0f;
  }
  __syncthreads();

  // discretize x tile -> h0 (f16). idx = clamp(ceil((v-min)*25/range)-1, 0, 24)
  // == searchsorted(side='left') on equal-width interior boundaries.
  for (int i = tid; i < MB * HID / 4; i += 256) {
    int r = i >> 7;           // / (HID/4)
    int c = (i & 127) << 2;
    float4 v = *(const float4*)(x + (size_t)(rowBase + r) * HID + c);
    float vv[4] = {v.x, v.y, v.z, v.w};
    h4_t q;
#pragma unroll
    for (int j = 0; j < 4; ++j) {
      float t = (vv[j] - cmn[c + j]) * csc[c + j];
      int idx = (int)ceilf(t) - 1;
      idx = idx < 0 ? 0 : (idx > (LEVELS - 1) ? (LEVELS - 1) : idx);
      q[j] = (_Float16)idx;
    }
    *(h4_t*)(h0 + r * LDH + c) = q;
  }
  __syncthreads();  // h0 complete before layer 0 compute

  _Float16* cur = h0;
  _Float16* nxt = h1;
  mlp_layer(cur, nxt, sB0, WtIn, bIn, HID, nullptr, rowBase);  // l_in
  { _Float16* t = cur; cur = nxt; nxt = t; }
#pragma unroll 1
  for (int l = 0; l < 5; ++l) {                                // l_h x5
    mlp_layer(cur, nxt, sB0, WtH, bH, HID, nullptr, rowBase);
    _Float16* t = cur; cur = nxt; nxt = t;
  }
  mlp_layer(cur, nxt, sB0, WtOut, bOut, DOUT, y, rowBase);     // l_out
}

// ---------------------------------------------------------------------------
extern "C" void kernel_launch(void* const* d_in, const int* in_sizes, int n_in,
                              void* d_out, int out_size, void* d_ws, size_t ws_size,
                              hipStream_t stream) {
  const float* x     = (const float*)d_in[0];
  const float* W_in  = (const float*)d_in[1];
  const float* b_in  = (const float*)d_in[2];
  const float* W_h   = (const float*)d_in[3];
  const float* b_h   = (const float*)d_in[4];
  const float* W_out = (const float*)d_in[5];
  const float* b_out = (const float*)d_in[6];
  float* y = (float*)d_out;

  // workspace: [umin 512][umax 512][pad->4KB][WtIn 512KB][WtH 512KB][WtOut 128KB]
  char* ws = (char*)d_ws;
  unsigned* umin  = (unsigned*)ws;
  unsigned* umax  = umin + HID;
  _Float16* WtIn  = (_Float16*)(ws + 4096);
  _Float16* WtH   = WtIn + (size_t)HID * HID;
  _Float16* WtOut = WtH + (size_t)HID * HID;

  init_minmax<<<1, 512, 0, stream>>>(umin, umax, y + (size_t)BTOT * DOUT);
  colminmax<<<BTOT / 256, 512, 0, stream>>>(x, umin, umax);
  convert_weights<<<256, 256, 0, stream>>>(W_in, WtIn, HID, HID);
  convert_weights<<<256, 256, 0, stream>>>(W_h, WtH, HID, HID);
  convert_weights<<<64, 256, 0, stream>>>(W_out, WtOut, HID, DOUT);

  // LDS: 2 weight panels + column (min,scale) + 2 h-buffers  (~300 KB < 320 KB)
  size_t smem = (size_t)(2 * MB * LDH + 2 * PANELH) * sizeof(_Float16)
              + (size_t)2 * HID * sizeof(float);
  hipFuncSetAttribute(reinterpret_cast<const void*>(fused_mlp),
                      hipFuncAttributeMaxDynamicSharedMemorySize, (int)smem);
  fused_mlp<<<NBLK, 256, smem, stream>>>(x, umin, umax, WtIn, WtH, WtOut,
                                         b_in, b_h, b_out, y);
}
